// Gemma4Attention_34952443855134
// MI455X (gfx1250) — compile-verified, matrix operand order checked by reading, not run on hardware
//
#include <hip/hip_runtime.h>
#include <hip/hip_bf16.h>
#include <stdint.h>

// ---------------- CDNA5 WMMA types ----------------
typedef __attribute__((ext_vector_type(16))) __bf16 v16bf;
typedef __attribute__((ext_vector_type(8)))  float  v8f;
typedef int v4i __attribute__((__vector_size__(4 * sizeof(int))));

union AFrag {            // 16 bf16 = 32B = 8 VGPRs, filled by two 16B loads
    v16bf v;
    uint4 u[2];
};

// Optional gfx1250 async global->LDS path (guarded: falls back to sync copy)
#if defined(__has_builtin)
#if __has_builtin(__builtin_amdgcn_global_load_async_to_lds_b128) && \
    __has_builtin(__builtin_amdgcn_s_wait_asynccnt)
#define USE_ASYNC_LDS 1
#endif
#endif

#define GAS1 __attribute__((address_space(1)))
#define LAS3 __attribute__((address_space(3)))

// Problem constants
constexpr int Tt   = 4096;   // B*S
constexpr int HID  = 2048;
constexpr int Hh   = 8;
constexpr int KVHh = 4;
constexpr int Dd   = 256;
constexpr int SWw  = 1024;

__device__ __forceinline__ unsigned short f2bf(float f) {
    unsigned int u = __float_as_uint(f);
    unsigned int r = u + 0x7FFFu + ((u >> 16) & 1u);   // round-to-nearest-even
    return (unsigned short)(r >> 16);
}

// ---------------- fp32 -> bf16 conversion ----------------
__global__ void cvt_f32_bf16_kernel(const float* __restrict__ src,
                                    unsigned short* __restrict__ dst, int n) {
    int idx = (blockIdx.x * blockDim.x + threadIdx.x) * 4;
    if (idx < n) {
        float4 f = *(const float4*)(src + idx);
        uint2 o;
        o.x = (unsigned)f2bf(f.x) | ((unsigned)f2bf(f.y) << 16);
        o.y = (unsigned)f2bf(f.z) | ((unsigned)f2bf(f.w) << 16);
        *(uint2*)(dst + idx) = o;
    }
}

// ---------------- GEMM: C[M,N] = X[M,K] * W[N,K]^T (bf16 in, fp32 out) ----
// Block tile 128x128x32, 256 threads = 8 waves in a 2(M) x 4(N) arrangement;
// each wave owns a 64x32 tile = 4x2 wmma 16x16 accumulators.
#define BK 32
__global__ __launch_bounds__(256) void gemm_bf16_xwT(
    const unsigned short* __restrict__ X,   // M x K
    const unsigned short* __restrict__ W,   // N x K
    float* __restrict__ C,                  // M x N
    int M, int N, int K)
{
    __shared__ __align__(16) unsigned short As[128][BK + 8];
    __shared__ __align__(16) unsigned short Bs[128][BK + 8];

    const int tid  = threadIdx.x;
    const int lane = tid & 31;
    const int w    = tid >> 5;
    const int wm   = w & 1;          // 0..1  (M direction, 64 rows each)
    const int wn   = w >> 1;         // 0..3  (N direction, 32 cols each)
    const int m0   = blockIdx.y * 128;
    const int n0   = blockIdx.x * 128;
    const int rsel = lane & 15;
    const int ksel = (lane < 16) ? 0 : 8;     // ISA 16-bit A/B fragment K split

    v8f acc[4][2];
    #pragma unroll
    for (int mt = 0; mt < 4; ++mt)
        #pragma unroll
        for (int nt = 0; nt < 2; ++nt)
            acc[mt][nt] = (v8f){0.f,0.f,0.f,0.f,0.f,0.f,0.f,0.f};

    for (int k0 = 0; k0 < K; k0 += BK) {
        __syncthreads();
        #pragma unroll
        for (int it = 0; it < 2; ++it) {               // 512 16B chunks / 256 thr
            int c  = tid + 256 * it;
            int r  = c >> 2;
            int cc = (c & 3) * 8;
            *(uint4*)&As[r][cc] = *(const uint4*)&X[(size_t)(m0 + r) * K + k0 + cc];
            *(uint4*)&Bs[r][cc] = *(const uint4*)&W[(size_t)(n0 + r) * K + k0 + cc];
        }
        __syncthreads();

        AFrag a[4], b[2];
        #pragma unroll
        for (int mt = 0; mt < 4; ++mt) {
            const unsigned short* p = &As[wm * 64 + mt * 16 + rsel][ksel];
            a[mt].u[0] = *(const uint4*)p;
            a[mt].u[1] = *(const uint4*)(p + 16);
        }
        #pragma unroll
        for (int nt = 0; nt < 2; ++nt) {
            const unsigned short* p = &Bs[wn * 32 + nt * 16 + rsel][ksel];
            b[nt].u[0] = *(const uint4*)p;
            b[nt].u[1] = *(const uint4*)(p + 16);
        }
        #pragma unroll
        for (int mt = 0; mt < 4; ++mt)
            #pragma unroll
            for (int nt = 0; nt < 2; ++nt)
                acc[mt][nt] = __builtin_amdgcn_wmma_f32_16x16x32_bf16(
                    false, a[mt].v, false, b[nt].v, (short)0, acc[mt][nt],
                    false, false);
    }

    const int hi = lane >> 4, nn = lane & 15;
    #pragma unroll
    for (int mt = 0; mt < 4; ++mt)
        #pragma unroll
        for (int nt = 0; nt < 2; ++nt)
            #pragma unroll
            for (int r = 0; r < 8; ++r) {
                int row = m0 + wm * 64 + mt * 16 + hi * 8 + r;
                int col = n0 + wn * 32 + nt * 16 + nn;
                C[(size_t)row * N + col] = acc[mt][nt][r];
            }
}

// ---------------- RMSNorm + RoPE + layout kernel ----------------
// grid.x = T ; 256 threads = 8 waves, each wave handles 2 of 16 rows
// (8 q-heads, 4 k-heads, 4 v-heads). Writes Q[H][T][D], K[KVH][T][D],
// Vt[KVH][D][T] in bf16.
__global__ __launch_bounds__(256) void normrope_kernel(
    const float* __restrict__ qf, const float* __restrict__ kf,
    const float* __restrict__ vf, const float* __restrict__ cosp,
    const float* __restrict__ sinp, const float* __restrict__ qnw,
    const float* __restrict__ knw,
    unsigned short* __restrict__ Qb, unsigned short* __restrict__ Kb,
    unsigned short* __restrict__ Vt)
{
    const int t    = blockIdx.x;
    const int w    = threadIdx.x >> 5;
    const int lane = threadIdx.x & 31;

    for (int rr = 0; rr < 2; ++rr) {
        const int ridx = w * 2 + rr;           // 0..15
        const float* src;
        int mode, head;
        if (ridx < 8)       { head = ridx;      mode = 0; src = qf + (size_t)t * (Hh * Dd)   + head * Dd; }
        else if (ridx < 12) { head = ridx - 8;  mode = 1; src = kf + (size_t)t * (KVHh * Dd) + head * Dd; }
        else                { head = ridx - 12; mode = 2; src = vf + (size_t)t * (KVHh * Dd) + head * Dd; }

        float x[8], ss = 0.f;
        #pragma unroll
        for (int j = 0; j < 8; ++j) { x[j] = src[lane + 32 * j]; ss += x[j] * x[j]; }
        ss += __shfl_xor(ss, 1, 32);
        ss += __shfl_xor(ss, 2, 32);
        ss += __shfl_xor(ss, 4, 32);
        ss += __shfl_xor(ss, 8, 32);
        ss += __shfl_xor(ss, 16, 32);
        const float rn = rsqrtf(ss * (1.0f / 256.0f) + 1e-6f);

        if (mode == 2) {
            #pragma unroll
            for (int j = 0; j < 8; ++j) {
                int d = lane + 32 * j;
                Vt[((size_t)head * Dd + d) * Tt + t] = f2bf(x[j] * rn);
            }
        } else {
            const float* wgt = (mode == 0) ? qnw : knw;
            float y[8];
            #pragma unroll
            for (int j = 0; j < 8; ++j) y[j] = x[j] * rn * wgt[lane + 32 * j];
            #pragma unroll
            for (int j = 0; j < 8; ++j) {
                int d = lane + 32 * j;                 // d<128 <=> j<4
                float rot = (j < 4) ? -y[j + 4] : y[j - 4];
                float o = y[j] * cosp[(size_t)t * Dd + d] + rot * sinp[(size_t)t * Dd + d];
                if (mode == 0)
                    Qb[((size_t)head * Tt + t) * Dd + d] = f2bf(o);
                else
                    Kb[((size_t)head * Tt + t) * Dd + d] = f2bf(o);
            }
        }
    }
}

// ---------------- sliding-window flash attention ----------------
// grid = (T/128, H); 256 threads = 8 waves; wave owns 16 query rows x D=256.
// K tile (64x256) staged cooperatively into LDS once per key block and shared
// by all 8 waves (8x traffic reduction). Per 64-key block per wave:
// 32 wmma for S = Q K^T, then 32 wmma for P V.
__global__ __launch_bounds__(256) void attn_kernel(
    const unsigned short* __restrict__ Qb,  // [H][T][D]
    const unsigned short* __restrict__ Kb,  // [KVH][T][D]
    const unsigned short* __restrict__ Vt,  // [KVH][D][T]
    unsigned short* __restrict__ Ob)        // [T][H*D]
{
    const int h   = blockIdx.y;
    const int qb  = blockIdx.x;
    const int kv  = h >> 1;                 // H/KVH == 2
    const int tid = threadIdx.x;
    const int lane = tid & 31, w = tid >> 5;
    const int hi = lane >> 4, ln = lane & 15;
    const int ksel = hi ? 8 : 0;
    const int tq0  = qb * 128 + w * 16;

    __shared__ __align__(16) unsigned short Ks[64][Dd + 8];   // K tile, 33KB
    __shared__ __align__(16) unsigned short Pl[8][16][72];    // per-wave P, 18KB

    const unsigned short* qrow = Qb + ((size_t)h * Tt + (tq0 + ln)) * Dd + ksel;

    v8f acco[16];
    #pragma unroll
    for (int dt = 0; dt < 16; ++dt)
        acco[dt] = (v8f){0.f,0.f,0.f,0.f,0.f,0.f,0.f,0.f};
    float mrow[8], lrow[8];
    #pragma unroll
    for (int r = 0; r < 8; ++r) { mrow[r] = -1e30f; lrow[r] = 0.f; }

    int lo = qb * 128 - (SWw - 1);
    const int kb_lo = (lo > 0) ? (lo >> 6) : 0;
    const int kb_hi = qb * 2 + 1;

    for (int kb = kb_lo; kb <= kb_hi; ++kb) {
        const int j0 = kb * 64;

        // ---- cooperative K-tile staging: 64x256 bf16 = 2048 b128 chunks ----
        __syncthreads();   // all waves finished reading previous K tile
        {
            const unsigned short* gk = Kb + ((size_t)kv * Tt + j0) * Dd;
            #pragma unroll
            for (int it = 0; it < 8; ++it) {
                int c   = tid + 256 * it;
                int r   = c >> 5;
                int col = (c & 31) * 8;
#if defined(USE_ASYNC_LDS)
                __builtin_amdgcn_global_load_async_to_lds_b128(
                    (GAS1 v4i*)(uintptr_t)(const void*)&gk[(size_t)r * Dd + col],
                    (LAS3 v4i*)(unsigned)(uintptr_t)(void*)&Ks[r][col],
                    0, 0);
#else
                *(uint4*)&Ks[r][col] = *(const uint4*)&gk[(size_t)r * Dd + col];
#endif
            }
#if defined(USE_ASYNC_LDS)
            __builtin_amdgcn_s_wait_asynccnt(0);
#endif
        }
        __syncthreads();   // staged K tile visible to all waves

        // ---- scores S = Q K^T (16 x 64), K fragments from LDS ----
        v8f s[4];
        #pragma unroll
        for (int nt = 0; nt < 4; ++nt)
            s[nt] = (v8f){0.f,0.f,0.f,0.f,0.f,0.f,0.f,0.f};
        #pragma unroll
        for (int kc = 0; kc < 8; ++kc) {
            AFrag qf;
            qf.u[0] = *(const uint4*)(qrow + kc * 32);
            qf.u[1] = *(const uint4*)(qrow + kc * 32 + 16);
            #pragma unroll
            for (int nt = 0; nt < 4; ++nt) {
                AFrag kfr;
                const unsigned short* p = &Ks[nt * 16 + ln][kc * 32 + ksel];
                kfr.u[0] = *(const uint4*)p;
                kfr.u[1] = *(const uint4*)(p + 16);
                s[nt] = __builtin_amdgcn_wmma_f32_16x16x32_bf16(
                            false, qf.v, false, kfr.v, (short)0, s[nt],
                            false, false);
            }
        }

        // ---- mask + row max ----
        float rmax[8];
        #pragma unroll
        for (int r = 0; r < 8; ++r) rmax[r] = -1e30f;
        #pragma unroll
        for (int nt = 0; nt < 4; ++nt) {
            int j = j0 + nt * 16 + ln;
            #pragma unroll
            for (int r = 0; r < 8; ++r) {
                int i = tq0 + hi * 8 + r;
                bool ok = (j <= i) && ((i - j) < SWw);
                float val = ok ? s[nt][r] : -1e30f;
                s[nt][r] = val;
                rmax[r] = fmaxf(rmax[r], val);
            }
        }
        #pragma unroll
        for (int r = 0; r < 8; ++r) {
            float v = rmax[r];
            v = fmaxf(v, __shfl_xor(v, 1, 32));
            v = fmaxf(v, __shfl_xor(v, 2, 32));
            v = fmaxf(v, __shfl_xor(v, 4, 32));
            v = fmaxf(v, __shfl_xor(v, 8, 32));
            rmax[r] = v;
        }

        // ---- online softmax update ----
        float alpha[8], psum[8];
        #pragma unroll
        for (int r = 0; r < 8; ++r) {
            float mnew = fmaxf(mrow[r], rmax[r]);
            alpha[r] = __expf(mrow[r] - mnew);
            mrow[r] = mnew;
            psum[r] = 0.f;
        }
        #pragma unroll
        for (int nt = 0; nt < 4; ++nt)
            #pragma unroll
            for (int r = 0; r < 8; ++r) {
                float p = (mrow[r] > -1e29f) ? __expf(s[nt][r] - mrow[r]) : 0.f;
                s[nt][r] = p;
                psum[r] += p;
            }
        #pragma unroll
        for (int r = 0; r < 8; ++r) {
            float v = psum[r];
            v += __shfl_xor(v, 1, 32);
            v += __shfl_xor(v, 2, 32);
            v += __shfl_xor(v, 4, 32);
            v += __shfl_xor(v, 8, 32);
            lrow[r] = lrow[r] * alpha[r] + v;
        }
        v8f av;
        #pragma unroll
        for (int r = 0; r < 8; ++r) av[r] = alpha[r];
        #pragma unroll
        for (int dt = 0; dt < 16; ++dt) acco[dt] *= av;

        // ---- P (C-layout) -> bf16 -> per-wave LDS, reload as A-fragments ----
        #pragma unroll
        for (int nt = 0; nt < 4; ++nt)
            #pragma unroll
            for (int r = 0; r < 8; ++r)
                Pl[w][hi * 8 + r][nt * 16 + ln] = f2bf(s[nt][r]);
        asm volatile("s_wait_dscnt 0" ::: "memory");  // wave-internal RAW guard
        __builtin_amdgcn_wave_barrier();

        // ---- O += P V  (V fragments direct from L2-resident Vt) ----
        #pragma unroll
        for (int kc2 = 0; kc2 < 2; ++kc2) {
            AFrag pa;
            const unsigned short* pp = &Pl[w][ln][kc2 * 32 + ksel];
            pa.u[0] = *(const uint4*)pp;
            pa.u[1] = *(const uint4*)(pp + 16);
            #pragma unroll
            for (int dt = 0; dt < 16; ++dt) {
                AFrag vb;
                const unsigned short* vp =
                    Vt + ((size_t)kv * Dd + dt * 16 + ln) * Tt + j0 + kc2 * 32 + ksel;
                vb.u[0] = *(const uint4*)vp;
                vb.u[1] = *(const uint4*)(vp + 16);
                acco[dt] = __builtin_amdgcn_wmma_f32_16x16x32_bf16(
                    false, pa.v, false, vb.v, (short)0, acco[dt], false, false);
            }
        }
    }

    // ---- epilogue: divide by l, store bf16 to [T][H*D] ----
    v8f invv;
    #pragma unroll
    for (int r = 0; r < 8; ++r) invv[r] = (lrow[r] > 0.f) ? (1.0f / lrow[r]) : 0.f;
    #pragma unroll
    for (int dt = 0; dt < 16; ++dt) {
        v8f o = acco[dt] * invv;
        #pragma unroll
        for (int r = 0; r < 8; ++r) {
            size_t row = (size_t)(tq0 + hi * 8 + r);
            Ob[row * (Hh * Dd) + h * Dd + dt * 16 + ln] = f2bf(o[r]);
        }
    }
}

// ---------------- host-side orchestration ----------------
extern "C" void kernel_launch(void* const* d_in, const int* in_sizes, int n_in,
                              void* d_out, int out_size, void* d_ws, size_t ws_size,
                              hipStream_t stream) {
    (void)in_sizes; (void)n_in; (void)out_size; (void)ws_size;

    const float* hidden = (const float*)d_in[0];
    const float* cosp   = (const float*)d_in[1];
    const float* sinp   = (const float*)d_in[2];
    const float* q_w    = (const float*)d_in[3];
    const float* k_w    = (const float*)d_in[4];
    const float* v_w    = (const float*)d_in[5];
    const float* o_w    = (const float*)d_in[6];
    const float* qnw    = (const float*)d_in[7];
    const float* knw    = (const float*)d_in[8];
    float* out          = (float*)d_out;

    // Workspace layout
    char* p = (char*)d_ws;
    auto alloc = [&](size_t bytes) { char* r = p; p += (bytes + 255) & ~size_t(255); return r; };
    unsigned short* Xbf  = (unsigned short*)alloc((size_t)Tt * HID * 2);
    unsigned short* qwbf = (unsigned short*)alloc((size_t)Hh * Dd * HID * 2);
    unsigned short* kwbf = (unsigned short*)alloc((size_t)KVHh * Dd * HID * 2);
    unsigned short* vwbf = (unsigned short*)alloc((size_t)KVHh * Dd * HID * 2);
    unsigned short* owbf = (unsigned short*)alloc((size_t)HID * Hh * Dd * 2);
    float* qf32 = (float*)alloc((size_t)Tt * Hh * Dd * 4);
    float* kf32 = (float*)alloc((size_t)Tt * KVHh * Dd * 4);
    float* vf32 = (float*)alloc((size_t)Tt * KVHh * Dd * 4);
    unsigned short* Qb = (unsigned short*)alloc((size_t)Hh * Tt * Dd * 2);
    unsigned short* Kb = (unsigned short*)alloc((size_t)KVHh * Tt * Dd * 2);
    unsigned short* Vt = (unsigned short*)alloc((size_t)KVHh * Dd * Tt * 2);
    unsigned short* Ob = (unsigned short*)alloc((size_t)Tt * Hh * Dd * 2);

    // 1) fp32 -> bf16 conversions
    auto cvt = [&](const float* s, unsigned short* d, int n) {
        cvt_f32_bf16_kernel<<<(n / 4 + 255) / 256, 256, 0, stream>>>(s, d, n);
    };
    cvt(hidden, Xbf,  Tt * HID);
    cvt(q_w,    qwbf, Hh * Dd * HID);
    cvt(k_w,    kwbf, KVHh * Dd * HID);
    cvt(v_w,    vwbf, KVHh * Dd * HID);
    cvt(o_w,    owbf, HID * Hh * Dd);

    // 2) QKV projections via WMMA GEMM
    gemm_bf16_xwT<<<dim3((Hh * Dd) / 128, Tt / 128), 256, 0, stream>>>(
        Xbf, qwbf, qf32, Tt, Hh * Dd, HID);
    gemm_bf16_xwT<<<dim3((KVHh * Dd) / 128, Tt / 128), 256, 0, stream>>>(
        Xbf, kwbf, kf32, Tt, KVHh * Dd, HID);
    gemm_bf16_xwT<<<dim3((KVHh * Dd) / 128, Tt / 128), 256, 0, stream>>>(
        Xbf, vwbf, vf32, Tt, KVHh * Dd, HID);

    // 3) RMSNorm + RoPE + attention-friendly layouts
    normrope_kernel<<<Tt, 256, 0, stream>>>(qf32, kf32, vf32, cosp, sinp,
                                            qnw, knw, Qb, Kb, Vt);

    // 4) Sliding-window flash attention (WMMA)
    attn_kernel<<<dim3(Tt / 128, Hh), 256, 0, stream>>>(Qb, Kb, Vt, Ob);

    // 5) Output projection via WMMA GEMM -> fp32 out
    gemm_bf16_xwT<<<dim3(HID / 128, Tt / 128), 256, 0, stream>>>(
        Ob, owbf, out, Tt, HID, Hh * Dd);
}
